// ODE_RNN_67997922230772
// MI455X (gfx1250) — matrix operand order repaired and checked by reading, not compiled, not run on hardware
//
#include <hip/hip_runtime.h>
#include <hip/hip_bf16.h>
#include <math.h>

// Problem constants (match reference)
#define Bb   64
#define Tt   4096
#define OBS  64
#define HID  25     // hidden size (pad to 32 for WMMA K/N)
#define WID  64     // MLP width
#define OUTD 64
#define GDIM 75     // 3*HID (GRU gates), pad to 80 (5 N-tiles)
#define INX  128    // GRU input = 2*OBS

typedef __attribute__((ext_vector_type(16))) _Float16 v16h;
typedef __attribute__((ext_vector_type(8)))  _Float16 f16x8;
typedef __attribute__((ext_vector_type(8)))  float    v8f;

// LDS strides (f16 units unless noted); all multiples of 8 halves => 16B rows
#define ASTR 72     // activation buffers [16][72], 64 used (odd-ish pad vs banks)
#define HSTR 40     // h-like f16 buffers [16][40], 32 used
#define XSTR 136    // x buffer [16][136], 128 used
#define GS   80     // gi/gh f32 stride

__device__ __forceinline__ float sigm(float x) {
    return 1.0f / (1.0f + __expf(-x));
}
__device__ __forceinline__ float tanh_fast(float x) {
    float t = __expf(-2.0f * fabsf(x));      // in (0,1], no overflow
    float r = (1.0f - t) / (1.0f + t);
    return __builtin_copysignf(r, x);
}

// ---- WMMA fragment helpers (layouts per CDNA5 ISA §7.12.2) ----

// A (16-bit, 16xK, K-chunk of 32): lane m=lane&15, hi=lane>>4.
// half j -> K = kbase+hi*16+j (contiguous, matches B fragments)
__device__ __forceinline__ v16h load_afrag(const _Float16* buf, int stride,
                                           int m, int kbase, int lhi) {
    const _Float16* p = buf + m * stride + kbase + lhi * 16;
    f16x8 lo = *(const f16x8*)p;
    f16x8 hi = *(const f16x8*)(p + 8);
    v16h a;
#pragma unroll
    for (int j = 0; j < 8; ++j) { a[j] = lo[j]; a[j + 8] = hi[j]; }
    return a;
}

// B (16-bit, 32x16 K-chunk): lane n=lane&15, hi=lane>>4; half j -> K = kbase+hi*16+j.
// Weights stored in LDS row-major as W[n][k] (B[k][n] = W[n][k] for z @ W.T),
// so a fragment is 16 contiguous f16 = two ds_load_b128.
__device__ __forceinline__ v16h load_bfrag_lds(const _Float16* W, int stride,
                                               int n, int kbase) {
    const _Float16* p = W + n * stride + kbase;
    f16x8 lo = *(const f16x8*)p;
    f16x8 hi = *(const f16x8*)(p + 8);
    v16h b;
#pragma unroll
    for (int j = 0; j < 8; ++j) { b[j] = lo[j]; b[j + 8] = hi[j]; }
    return b;
}

// Build a B fragment in registers directly from global f32 W[Nout][Kin],
// zero-padding out-of-range rows/cols (keeps WMMA pad lanes exactly zero).
__device__ __forceinline__ v16h make_bfrag(const float* W, int Nout, int Kin,
                                           int n, int kbase) {
    v16h b;
#pragma unroll
    for (int j = 0; j < 16; ++j) {
        int k = kbase + j;
        float v = (n < Nout && k < Kin) ? W[n * Kin + k] : 0.0f;
        b[j] = (_Float16)v;
    }
    return b;
}

__device__ __forceinline__ v8f wmma_f16(v16h a, v16h b, v8f c) {
    // D = A(16x32 f16) x B(32x16 f16) + C(16x16 f32)
    return __builtin_amdgcn_wmma_f32_16x16x32_f16(
        /*neg_a=*/false, a, /*neg_b=*/false, b,
        /*c_mod=*/(short)0, c, /*reuse_a=*/false, /*reuse_b=*/false);
}

__global__ __launch_bounds__(128, 1)
void odernn_persistent(const float* __restrict__ ts, const float* __restrict__ ys,
                       const float* __restrict__ W_ih, const float* __restrict__ b_ih,
                       const float* __restrict__ W_hh, const float* __restrict__ b_hh,
                       const float* __restrict__ Wm0, const float* __restrict__ Wm1,
                       const float* __restrict__ Wm2, const float* __restrict__ Wm3,
                       const float* __restrict__ Wm4,
                       const float* __restrict__ bm0, const float* __restrict__ bm1,
                       const float* __restrict__ bm2, const float* __restrict__ bm3,
                       const float* __restrict__ bm4,
                       const float* __restrict__ W_dec, const float* __restrict__ b_dec,
                       float* __restrict__ out) {
    __shared__ alignas(16) _Float16 actA[16 * ASTR];     // MLP activations ping
    __shared__ alignas(16) _Float16 actB[16 * ASTR];     // MLP activations pong
    __shared__ alignas(16) _Float16 hx16[16 * HSTR];     // f16 view of current A-input (h / h_mid / h_ode / h_new)
    __shared__ alignas(16) _Float16 x16 [16 * XSTR];     // GRU input [ys~ , mask] f16
    __shared__ alignas(16) _Float16 wih_l[80 * 128];     // W_ih f16, zero-padded rows
    __shared__ alignas(16) _Float16 whh_l[80 * 32];      // W_hh f16, zero-padded
    __shared__ float hbuf   [16 * 32];                   // f32 recurrent state (pad cols zero)
    __shared__ float kbuf   [16 * 32];                   // vf output k1/k2
    __shared__ float hodebuf[16 * 32];
    __shared__ float gibuf  [16 * GS];                   // x-side gate preacts (+b_ih)
    __shared__ float ghbuf  [16 * GS];                   // h-side gate preacts (+b_hh)
    __shared__ float dtbuf  [16];

    const int tid   = threadIdx.x;
    const int wave  = tid >> 5;      // wave32
    const int lane  = tid & 31;
    const int lrow  = lane & 15;
    const int lhi   = lane >> 4;
    const int rbase = blockIdx.x * 16;           // batch-row tile
    const int ncol  = wave * 16 + lrow;          // this lane's output column (N)

    // ---------------- one-time init ----------------
    for (int i = tid; i < 80 * 128; i += 128) {
        int n = i >> 7, k = i & 127;
        wih_l[i] = (_Float16)((n < GDIM) ? W_ih[n * INX + k] : 0.0f);
    }
    for (int i = tid; i < 80 * 32; i += 128) {
        int n = i >> 5, k = i & 31;
        whh_l[i] = (_Float16)((n < GDIM && k < HID) ? W_hh[n * HID + k] : 0.0f);
    }
    for (int i = tid; i < 16 * 32; i += 128) { hbuf[i] = 0.f; hodebuf[i] = 0.f; kbuf[i] = 0.f; }
    for (int i = tid; i < 16 * HSTR; i += 128) hx16[i] = (_Float16)0.0f;   // h0 = 0

    // Register-resident B fragments for MLP + decode (constant over the scan)
    v16h wb0 = make_bfrag(Wm0, WID, HID, ncol, lhi * 16);              // 25->64, K-pad 32
    const float* Wls[3] = { Wm1, Wm2, Wm3 };
    v16h wb1[3][2];
#pragma unroll
    for (int l = 0; l < 3; ++l)
#pragma unroll
        for (int kc = 0; kc < 2; ++kc)
            wb1[l][kc] = make_bfrag(Wls[l], WID, WID, ncol, kc * 32 + lhi * 16);
    v16h wb4[2];                                                       // 64->25 (waves 0,1 use)
#pragma unroll
    for (int kc = 0; kc < 2; ++kc)
        wb4[kc] = make_bfrag(Wm4, HID, WID, ncol, kc * 32 + lhi * 16);
    v16h wbdec = make_bfrag(W_dec, OUTD, HID, ncol, lhi * 16);         // 25->64

    // Per-lane bias registers (zero in pad lanes)
    float b0r = bm0[ncol];
    float b123r[3] = { bm1[ncol], bm2[ncol], bm3[ncol] };
    float b4r = (ncol < HID) ? bm4[ncol] : 0.0f;
    const int ng0 = wave * 16 + lrow;       // GRU gate tile owned by every wave
    const int ng1 = 64 + lrow;              // 5th gate tile (wave 3 only)
    float bihr0 = (ng0 < GDIM) ? b_ih[ng0] : 0.0f;
    float bhhr0 = (ng0 < GDIM) ? b_hh[ng0] : 0.0f;
    float bihr1 = (ng1 < GDIM) ? b_ih[ng1] : 0.0f;
    float bhhr1 = (ng1 < GDIM) ? b_hh[ng1] : 0.0f;
    float bdecr = b_dec[ncol];

    // out[:, 0, :] = ys[:, 0, :] (raw, per reference)
#pragma unroll
    for (int i = 0; i < 8; ++i) {
        int idx = tid + i * 128;
        int r = idx >> 6, c = idx & 63;
        out[((size_t)(rbase + r) * Tt) * OUTD + c] =
            ys[((size_t)(rbase + r) * Tt) * OBS + c];
    }
    __syncthreads();

    // vf: reads hx16 (K=32), writes kbuf[16][32]; per-wave N-tile, LDS ping-pong
    auto run_vf = [&]() {
        {   // L0: 25(pad32) -> 64
            v16h a = load_afrag(hx16, HSTR, lrow, 0, lhi);
            v8f c = {};
            c = wmma_f16(a, wb0, c);
#pragma unroll
            for (int v = 0; v < 8; ++v)
                actA[(v + lhi * 8) * ASTR + ncol] = (_Float16)tanh_fast(c[v] + b0r);
        }
        __syncthreads();
        _Float16* src = actA;
        _Float16* dst = actB;
#pragma unroll
        for (int l = 0; l < 3; ++l) {       // L1..L3: 64 -> 64
            v8f c = {};
#pragma unroll
            for (int kc = 0; kc < 2; ++kc) {
                v16h a = load_afrag(src, ASTR, lrow, kc * 32, lhi);
                c = wmma_f16(a, wb1[l][kc], c);
            }
#pragma unroll
            for (int v = 0; v < 8; ++v)
                dst[(v + lhi * 8) * ASTR + ncol] = (_Float16)tanh_fast(c[v] + b123r[l]);
            __syncthreads();
            _Float16* tmp = src; src = dst; dst = tmp;
        }
        if (wave < 2) {                     // L4: 64 -> 25(pad32), linear  (final act in actB)
            v8f c = {};
#pragma unroll
            for (int kc = 0; kc < 2; ++kc) {
                v16h a = load_afrag(actB, ASTR, lrow, kc * 32, lhi);
                c = wmma_f16(a, wb4[kc], c);
            }
#pragma unroll
            for (int v = 0; v < 8; ++v)
                kbuf[(v + lhi * 8) * 32 + ncol] = c[v] + b4r;   // pad cols write exact 0
        }
        __syncthreads();
    };

    // ---------------- sequential scan ----------------
#pragma unroll 1
    for (int t = 1; t < Tt; ++t) {
        // Issue x_t global loads now; consumed ~10 barriers later (latency hidden)
        float ysv[8];
#pragma unroll
        for (int i = 0; i < 8; ++i) {
            int idx = tid + i * 128;
            int r = idx >> 6, c = idx & 63;
            ysv[i] = ys[((size_t)(rbase + r) * Tt + t) * OBS + c];
        }
        if ((t + 1) < Tt && (tid & 7) == 0)
            __builtin_prefetch(&ys[((size_t)(rbase + (tid >> 3)) * Tt + (t + 1)) * OBS], 0, 0);
        if (tid < 16)
            dtbuf[tid] = ts[(size_t)(rbase + tid) * Tt + t]
                       - ts[(size_t)(rbase + tid) * Tt + (t - 1)];

        run_vf();                                   // k1 -> kbuf

        // h_mid = h + 0.5*dt*k1  (f16 into hx16)
#pragma unroll
        for (int i = 0; i < 4; ++i) {
            int idx = tid + i * 128;
            int r = idx >> 5, c = idx & 31;
            float hm = hbuf[idx] + 0.5f * dtbuf[r] * kbuf[idx];
            hx16[r * HSTR + c] = (_Float16)hm;
        }
        __syncthreads();

        run_vf();                                   // k2 -> kbuf

        // h_ode = h + dt*k2 ; also stage x_t into LDS (nan_to_num + mask)
#pragma unroll
        for (int i = 0; i < 4; ++i) {
            int idx = tid + i * 128;
            int r = idx >> 5, c = idx & 31;
            float ho = hbuf[idx] + dtbuf[r] * kbuf[idx];
            hodebuf[idx] = ho;
            hx16[r * HSTR + c] = (_Float16)ho;
        }
#pragma unroll
        for (int i = 0; i < 8; ++i) {
            int idx = tid + i * 128;
            int r = idx >> 6, c = idx & 63;
            float v = ysv[i];
            bool ok = (v == v);                     // !isnan
            x16[r * XSTR + c]      = (_Float16)(ok ? v : 0.0f);
            x16[r * XSTR + 64 + c] = (_Float16)(ok ? 1.0f : 0.0f);
        }
        __syncthreads();

        // GRU pre-activations: gi = x@W_ih.T (K=128), gh = h_ode@W_hh.T (K=32)
        // 5 N-tiles of 16 over 75(pad80) gates; wave w -> tile w, wave 3 also tile 4.
#pragma unroll
        for (int tt = 0; tt < 2; ++tt) {
            if (tt == 0 || wave == 3) {             // wave-uniform: EXEC stays all-1
                int   n  = (tt == 0) ? ng0 : ng1;
                float bi = (tt == 0) ? bihr0 : bihr1;
                float bh = (tt == 0) ? bhhr0 : bhhr1;
                v8f ci = {};
#pragma unroll
                for (int kc = 0; kc < 4; ++kc) {
                    v16h a = load_afrag(x16, XSTR, lrow, kc * 32, lhi);
                    v16h b = load_bfrag_lds(wih_l, 128, n, kc * 32 + lhi * 16);
                    ci = wmma_f16(a, b, ci);
                }
                v16h ah  = load_afrag(hx16, HSTR, lrow, 0, lhi);
                v16h bhf = load_bfrag_lds(whh_l, 32, n, lhi * 16);
                v8f ch = {};
                ch = wmma_f16(ah, bhf, ch);
#pragma unroll
                for (int v = 0; v < 8; ++v) {
                    int m = v + lhi * 8;
                    gibuf[m * GS + n] = ci[v] + bi;
                    ghbuf[m * GS + n] = ch[v] + bh;
                }
            }
        }
        __syncthreads();

        // Gates: r=sig(i_r+h_r), z=sig(i_z+h_z), n=tanh(i_n + r*h_n),
        //        h = (1-z)*n + z*h_ode
#pragma unroll
        for (int i = 0; i < 4; ++i) {
            int idx = tid + i * 128;
            int r = idx >> 5, c = idx & 31;
            if (c < HID) {
                float gr  = gibuf[r * GS + c]           + ghbuf[r * GS + c];
                float gz  = gibuf[r * GS + HID + c]     + ghbuf[r * GS + HID + c];
                float gin = gibuf[r * GS + 2 * HID + c];
                float ghn = ghbuf[r * GS + 2 * HID + c];
                float rg = sigm(gr);
                float zg = sigm(gz);
                float nn = tanh_fast(gin + rg * ghn);
                float hnew = (1.0f - zg) * nn + zg * hodebuf[idx];
                hbuf[idx] = hnew;
                hx16[r * HSTR + c] = (_Float16)hnew;
            }
        }
        __syncthreads();

        // Decode: out[:, t, :] = h_new @ W_dec.T + b_dec (each wave its N-tile)
        {
            v16h a = load_afrag(hx16, HSTR, lrow, 0, lhi);
            v8f c = {};
            c = wmma_f16(a, wbdec, c);
#pragma unroll
            for (int v = 0; v < 8; ++v) {
                int m = v + lhi * 8;
                out[((size_t)(rbase + m) * Tt + t) * OUTD + ncol] = c[v] + bdecr;
            }
        }
        // No barrier needed: hx16 is only re-read (L0) until the next h_mid write,
        // which is separated by 5 barriers.
    }
}

extern "C" void kernel_launch(void* const* d_in, const int* in_sizes, int n_in,
                              void* d_out, int out_size, void* d_ws, size_t ws_size,
                              hipStream_t stream) {
    (void)in_sizes; (void)n_in; (void)out_size; (void)d_ws; (void)ws_size;
    const float* ts    = (const float*)d_in[0];
    const float* ys    = (const float*)d_in[1];
    const float* W_ih  = (const float*)d_in[2];
    const float* b_ih  = (const float*)d_in[3];
    const float* W_hh  = (const float*)d_in[4];
    const float* b_hh  = (const float*)d_in[5];
    const float* Wm0   = (const float*)d_in[6];
    const float* Wm1   = (const float*)d_in[7];
    const float* Wm2   = (const float*)d_in[8];
    const float* Wm3   = (const float*)d_in[9];
    const float* Wm4   = (const float*)d_in[10];
    const float* bm0   = (const float*)d_in[11];
    const float* bm1   = (const float*)d_in[12];
    const float* bm2   = (const float*)d_in[13];
    const float* bm3   = (const float*)d_in[14];
    const float* bm4   = (const float*)d_in[15];
    const float* W_dec = (const float*)d_in[16];
    const float* b_dec = (const float*)d_in[17];
    float* out = (float*)d_out;

    odernn_persistent<<<dim3(Bb / 16), dim3(128), 0, stream>>>(
        ts, ys, W_ih, b_ih, W_hh, b_hh,
        Wm0, Wm1, Wm2, Wm3, Wm4, bm0, bm1, bm2, bm3, bm4,
        W_dec, b_dec, out);
}
